// GemmaAttention_15255723835665
// MI455X (gfx1250) — matrix operand order repaired and checked by reading, not compiled, not run on hardware
//
#include <hip/hip_runtime.h>
#include <hip/hip_bf16.h>
#include <math.h>

// ---------------------------------------------------------------------------
// Problem constants (match reference)
// ---------------------------------------------------------------------------
namespace cfg {
constexpr int B    = 4;
constexpr int Q    = 1024;
constexpr int HID  = 2048;   // HIDDEN
constexpr int NH   = 8;
constexpr int HD   = 256;
constexpr int KV   = 256;    // NKV * HD
}

typedef __attribute__((ext_vector_type(16))) _Float16 v16h;
typedef __attribute__((ext_vector_type(8)))  _Float16 h8v;
typedef __attribute__((ext_vector_type(8)))  float    v8f;

// ---------------------------------------------------------------------------
// WMMA fragment loaders per CDNA5 ISA 7.12.2 (wave32)
// A 16x32 f16: lane holds row M=lane&15; K = (lane<16 ? {0..7,16..23}
//                                                      : {8..15,24..31})
// ---------------------------------------------------------------------------
__device__ __forceinline__ v16h load_a_frag(const _Float16* __restrict__ A,
                                            int lda, int row, int k0) {
  const int lane = threadIdx.x & 31;
  const int r    = row + (lane & 15);
  const int kb   = (lane >> 4) << 3;               // 0 or 8
  const _Float16* p = A + (size_t)r * lda + (k0 + kb);
  h8v lo = *(const h8v*)(p);                       // K = k0+kb    .. +7
  h8v hi = *(const h8v*)(p + 16);                  // K = k0+kb+16 .. +23
  v16h f;
#pragma unroll
  for (int i = 0; i < 8; ++i) { f[i] = lo[i]; f[i + 8] = hi[i]; }
  return f;
}

// B 32x16 f16, operand stored as (N,K) row-major (K-contiguous, i.e. W for
// x @ W^T):  lane holds col N=lane&15; K = (lane<16 ? {0..7,16..23}
//                                                      : {8..15,24..31})
__device__ __forceinline__ v16h load_b_frag_bt(const _Float16* __restrict__ Bt,
                                               int ldb, int col, int k0) {
  const int lane = threadIdx.x & 31;
  const int n    = col + (lane & 15);
  const int kb   = (lane >> 4) << 3;               // 0 or 8
  const _Float16* p = Bt + (size_t)n * ldb + (k0 + kb);
  h8v lo = *(const h8v*)(p);
  h8v hi = *(const h8v*)(p + 16);
  v16h f;
#pragma unroll
  for (int i = 0; i < 8; ++i) { f[i] = lo[i]; f[i + 8] = hi[i]; }
  return f;
}

// C/D 16x16 f32: VGPR r holds row r + 8*(lane>=16), col = lane&15
__device__ __forceinline__ void store_tile(float* __restrict__ C, int ldc,
                                           int row, int col, v8f acc,
                                           float scale,
                                           const float* __restrict__ mask,
                                           int ldm) {
  const int lane = threadIdx.x & 31;
  const int n    = col + (lane & 15);
  const int rb   = row + ((lane >> 4) << 3);
#pragma unroll
  for (int r = 0; r < 8; ++r) {
    float v = acc[r] * scale;
    if (mask) v += mask[(size_t)(rb + r) * ldm + n];
    C[(size_t)(rb + r) * ldc + n] = v;
  }
}

#define WMMA_F16(a, b, c) \
  __builtin_amdgcn_wmma_f32_16x16x32_f16(false, (a), false, (b), (short)0, (c), false, false)

// ---------------------------------------------------------------------------
// WMMA GEMM:  C[M,N](f32) = scale * A[M,K](f16) x Bt[N,K](f16)^T + mask
// blockDim = 256 (8 waves, 4 in M x 2 in N). Block tile 128x128.
// Wave tile 32x64 = 2x4 WMMA tiles -> 8 WMMAs per 12 b128 loads per K-step.
// grid.z indexes (batch*heads); per-z operand offsets via strides.
// ---------------------------------------------------------------------------
__global__ void __launch_bounds__(256)
wmma_gemm(const _Float16* __restrict__ A, int lda, long long aB, long long aH,
          const _Float16* __restrict__ Bt, int ldb, long long bB, long long bH,
          float* __restrict__ C, int ldc, long long cB, long long cH,
          int K, float scale,
          const float* __restrict__ mask, long long maskB, int ldm, int nh) {
  const int z = blockIdx.z;
  const int b = z / nh;
  const int h = z % nh;
  A  += (size_t)b * aB + (size_t)h * aH;
  Bt += (size_t)b * bB + (size_t)h * bH;
  C  += (size_t)b * cB + (size_t)h * cH;
  if (mask) mask += (size_t)b * maskB;

  const int w  = threadIdx.x >> 5;
  const int m0 = blockIdx.x * 128 + (w & 3) * 32;   // wave covers rows m0..m0+31
  const int n0 = blockIdx.y * 128 + (w >> 2) * 64;  // wave covers cols n0..n0+63

  v8f acc[2][4] = {};
  for (int k0 = 0; k0 < K; k0 += 32) {
    v16h a[2];
#pragma unroll
    for (int i = 0; i < 2; ++i) a[i] = load_a_frag(A, lda, m0 + 16 * i, k0);
    v16h bf[4];
#pragma unroll
    for (int j = 0; j < 4; ++j) bf[j] = load_b_frag_bt(Bt, ldb, n0 + 16 * j, k0);
#pragma unroll
    for (int i = 0; i < 2; ++i)
#pragma unroll
      for (int j = 0; j < 4; ++j)
        acc[i][j] = WMMA_F16(a[i], bf[j], acc[i][j]);
  }
#pragma unroll
  for (int i = 0; i < 2; ++i)
#pragma unroll
    for (int j = 0; j < 4; ++j)
      store_tile(C, ldc, m0 + 16 * i, n0 + 16 * j, acc[i][j], scale, mask, ldm);
}

// ---------------------------------------------------------------------------
// Elementwise / reshape helpers
// ---------------------------------------------------------------------------
__global__ void f32_to_f16_kernel(const float* __restrict__ x,
                                  _Float16* __restrict__ y, size_t n) {
  size_t i = (size_t)blockIdx.x * blockDim.x + threadIdx.x;
  if (i < n) y[i] = (_Float16)x[i];
}

// Fused f32->f16 transpose: X (B*Qn rows, C cols) -> Y (B, C, Qn).
// LDS-tiled 32x32 so both global read and write are coalesced.
__global__ void __launch_bounds__(256)
transpose_cvt_kernel(const float* __restrict__ X, _Float16* __restrict__ Y,
                     int Qn, int C) {
  __shared__ float tile[32][33];
  const int b  = blockIdx.z;
  const int c0 = blockIdx.x * 32;
  const int q0 = blockIdx.y * 32;
  const int tx = threadIdx.x & 31;
  const int ty = threadIdx.x >> 5;          // 0..7
#pragma unroll
  for (int j = 0; j < 4; ++j)
    tile[ty + 8 * j][tx] =
        X[((size_t)b * Qn + q0 + ty + 8 * j) * C + c0 + tx];
  __syncthreads();
#pragma unroll
  for (int j = 0; j < 4; ++j)
    Y[(size_t)b * C * Qn + (size_t)(c0 + ty + 8 * j) * Qn + q0 + tx] =
        (_Float16)tile[tx][ty + 8 * j];
}

// RoPE (Gemma rotate-half), reads f32 projection, writes f16.
// width = NH*HD (q) or KV (k); rows = B*Q. pos taken from position_ids.
__global__ void rope_kernel(const float* __restrict__ X,
                            _Float16* __restrict__ Y,
                            const int* __restrict__ pos_ids,
                            int width, size_t total) {
  size_t idx = (size_t)blockIdx.x * blockDim.x + threadIdx.x;
  if (idx >= total) return;
  const int    col = (int)(idx % width);
  const size_t row = idx / width;
  const int d = col & (cfg::HD - 1);        // position within the head
  const int j = d & (cfg::HD / 2 - 1);      // freq index (d % 128)
  const float pos = (float)pos_ids[row];
  const float inv = __powf(10000.0f, -(float)j * (2.0f / (float)cfg::HD));
  const float ang = pos * inv;
  const float c = cosf(ang), s = sinf(ang);
  const float x = X[idx];
  const float rot = (d < cfg::HD / 2) ? -X[idx + cfg::HD / 2]
                                      :  X[idx - cfg::HD / 2];
  Y[idx] = (_Float16)(x * c + rot * s);
}

// Row softmax over n=1024 columns. In-place on f32 (d_out attn_weights slice),
// also emits f16 probabilities for the P@V WMMA GEMM.
__global__ void __launch_bounds__(256)
softmax_kernel(float* __restrict__ S, _Float16* __restrict__ P16, int n) {
  __shared__ float red[256];
  const size_t row = blockIdx.x;
  float* r = S + row * (size_t)n;
  const int t = threadIdx.x;
  float vals[4];
  float lmax = -INFINITY;
#pragma unroll
  for (int i = 0; i < 4; ++i) {
    vals[i] = r[t + 256 * i];
    lmax = fmaxf(lmax, vals[i]);
  }
  red[t] = lmax;
  __syncthreads();
  for (int s = 128; s > 0; s >>= 1) {
    if (t < s) red[t] = fmaxf(red[t], red[t + s]);
    __syncthreads();
  }
  const float m = red[0];
  __syncthreads();
  float lsum = 0.f;
#pragma unroll
  for (int i = 0; i < 4; ++i) {
    vals[i] = expf(vals[i] - m);
    lsum += vals[i];
  }
  red[t] = lsum;
  __syncthreads();
  for (int s = 128; s > 0; s >>= 1) {
    if (t < s) red[t] += red[t + s];
    __syncthreads();
  }
  const float inv = 1.0f / red[0];
#pragma unroll
  for (int i = 0; i < 4; ++i) {
    const float v = vals[i] * inv;
    r[t + 256 * i] = v;
    P16[row * (size_t)n + t + 256 * i] = (_Float16)v;
  }
}

// ---------------------------------------------------------------------------
// Host-side orchestration
// ---------------------------------------------------------------------------
extern "C" void kernel_launch(void* const* d_in, const int* in_sizes, int n_in,
                              void* d_out, int out_size, void* d_ws,
                              size_t ws_size, hipStream_t stream) {
  (void)in_sizes; (void)n_in; (void)out_size; (void)ws_size;
  using namespace cfg;

  const float* hs  = (const float*)d_in[0];   // (B,Q,HID)
  const float* msk = (const float*)d_in[1];   // (B,1,Q,Q)
  const int*   pos = (const int*)  d_in[2];   // (B,Q)
  const float* wq  = (const float*)d_in[3];   // (NH*HD, HID)
  const float* wk  = (const float*)d_in[4];   // (KV, HID)
  const float* wv  = (const float*)d_in[5];   // (KV, HID)
  const float* wo  = (const float*)d_in[6];   // (HID, NH*HD)

  float* out_attn = (float*)d_out;                       // (B,Q,HID)
  float* out_w    = out_attn + (size_t)B * Q * HID;      // (B,NH,Q,Q)

  // --- carve workspace ----------------------------------------------------
  char* cur = (char*)d_ws;
  auto carve = [&](size_t bytes) -> void* {
    void* p = (void*)cur;
    cur += (bytes + 255) & ~(size_t)255;
    return p;
  };
  const size_t BQ = (size_t)B * Q;
  _Float16* hs16 = (_Float16*)carve(BQ * HID * 2);
  _Float16* wq16 = (_Float16*)carve((size_t)NH * HD * HID * 2);
  _Float16* wk16 = (_Float16*)carve((size_t)KV * HID * 2);
  _Float16* wv16 = (_Float16*)carve((size_t)KV * HID * 2);
  _Float16* wo16 = (_Float16*)carve((size_t)HID * NH * HD * 2);
  _Float16* q16  = (_Float16*)carve(BQ * NH * HD * 2);
  _Float16* k16  = (_Float16*)carve(BQ * KV * 2);
  _Float16* vt16 = (_Float16*)carve(BQ * KV * 2);       // (B, KV, Q) transposed
  _Float16* p16  = (_Float16*)carve((size_t)B * NH * Q * Q * 2);
  _Float16* ao16 = (_Float16*)carve(BQ * HID * 2);
  float*    qf   = (float*)carve(BQ * NH * HD * 4);
  float*    kf   = (float*)carve(BQ * KV * 4);
  float*    vf   = (float*)carve(BQ * KV * 4);
  float*    aof  = (float*)carve(BQ * HID * 4);

  const dim3 blk(256);
  auto cvt = [&](const float* x, _Float16* y, size_t n) {
    f32_to_f16_kernel<<<dim3((unsigned)((n + 255) / 256)), blk, 0, stream>>>(x, y, n);
  };

  // 1) f32 -> f16 conversions of activations and weights
  cvt(hs, hs16, BQ * HID);
  cvt(wq, wq16, (size_t)NH * HD * HID);
  cvt(wk, wk16, (size_t)KV * HID);
  cvt(wv, wv16, (size_t)KV * HID);
  cvt(wo, wo16, (size_t)HID * NH * HD);

  // 2) Projections: X(BQ,HID) @ W^T  -> f32
  wmma_gemm<<<dim3(BQ / 128, (NH * HD) / 128, 1), blk, 0, stream>>>(
      hs16, HID, 0, 0, wq16, HID, 0, 0, qf, NH * HD, 0, 0,
      HID, 1.0f, nullptr, 0, 0, 1);
  wmma_gemm<<<dim3(BQ / 128, KV / 128, 1), blk, 0, stream>>>(
      hs16, HID, 0, 0, wk16, HID, 0, 0, kf, KV, 0, 0,
      HID, 1.0f, nullptr, 0, 0, 1);
  wmma_gemm<<<dim3(BQ / 128, KV / 128, 1), blk, 0, stream>>>(
      hs16, HID, 0, 0, wv16, HID, 0, 0, vf, KV, 0, 0,
      HID, 1.0f, nullptr, 0, 0, 1);

  // 3) RoPE on q and k (f32 -> f16); V gets a fused f32->f16 transpose into
  //    (B, KV, Q) so the P@V GEMM can use the K-contiguous b128 loader.
  {
    size_t nq = BQ * NH * HD;
    rope_kernel<<<dim3((unsigned)((nq + 255) / 256)), blk, 0, stream>>>(
        qf, q16, pos, NH * HD, nq);
    size_t nk = BQ * KV;
    rope_kernel<<<dim3((unsigned)((nk + 255) / 256)), blk, 0, stream>>>(
        kf, k16, pos, KV, nk);
    transpose_cvt_kernel<<<dim3(KV / 32, Q / 32, B), blk, 0, stream>>>(
        vf, vt16, Q, KV);
  }

  // 4) scores = Q @ K^T / sqrt(HD) + mask  -> f32 straight into d_out weights
  wmma_gemm<<<dim3(Q / 128, Q / 128, B * NH), blk, 0, stream>>>(
      q16, NH * HD, (long long)Q * NH * HD, (long long)HD,
      k16, KV,      (long long)Q * KV,      0,
      out_w, Q,     (long long)NH * Q * Q,  (long long)Q * Q,
      HD, 1.0f / 16.0f /* 1/sqrt(256) */,
      msk, (long long)Q * Q, Q, NH);

  // 5) softmax rows (in-place f32, plus f16 copy for P@V)
  softmax_kernel<<<dim3((unsigned)((size_t)B * NH * Q)), blk, 0, stream>>>(
      out_w, p16, Q);

  // 6) attn_out = P @ Vt^T   (Vt is (B, KV, Q), K-contiguous)
  wmma_gemm<<<dim3(Q / 128, HD / 128, B * NH), blk, 0, stream>>>(
      p16, Q,   (long long)NH * Q * Q, (long long)Q * Q,
      vt16, Q,  (long long)KV * Q,     0,
      aof, NH * HD, (long long)Q * NH * HD, (long long)HD,
      Q, 1.0f, nullptr, 0, 0, NH);
  cvt(aof, ao16, BQ * HID);

  // 7) final projection: attn_out @ w_o^T -> d_out attn_output
  wmma_gemm<<<dim3(BQ / 128, HID / 128, 1), blk, 0, stream>>>(
      ao16, HID, 0, 0, wo16, NH * HD, 0, 0, out_attn, HID, 0, 0,
      NH * HD, 1.0f, nullptr, 0, 0, 1);
}
